// DMMGNN_7464653161135
// MI455X (gfx1250) — matrix operand order repaired and checked by reading, not compiled, not run on hardware
//
#include <hip/hip_runtime.h>
#include <hip/hip_bf16.h>
#include <cstdint>

#define DEVI __device__ __forceinline__

typedef __attribute__((ext_vector_type(16))) __bf16 v16bf;
typedef __attribute__((ext_vector_type(8)))  float  v8f;

union FragU { v16bf v; uint4 u[2]; };

DEVI __bf16 f2bf(float f) {
  unsigned u = __builtin_bit_cast(unsigned, f);
  unsigned r = (u + 0x7FFFu + ((u >> 16) & 1u)) >> 16;
  return __builtin_bit_cast(__bf16, (unsigned short)r);
}
DEVI float bf2f(__bf16 b) {
  unsigned u = ((unsigned)__builtin_bit_cast(unsigned short, b)) << 16;
  return __builtin_bit_cast(float, u);
}
DEVI float geluf(float x) { return 0.5f * x * (1.0f + erff(x * 0.7071067811865475f)); }

// CDNA5 16-bit A fragment: lane<16 -> K {c..c+7, c+16..c+23}, lane>=16 adds +8.
// Caller passes p = rowbase + kk + 8*(lane>>4).
DEVI v16bf load_fragA(const __bf16* p) {
  FragU f;
  f.u[0] = *reinterpret_cast<const uint4*>(p);
  f.u[1] = *reinterpret_cast<const uint4*>(p + 16);
  return f.v;
}
// CDNA5 16-bit B fragment: lane<16 -> K {c..c+7, c+16..c+23}, lane>=16 adds +8.
// Caller passes p = colbase + kk + 8*(lane>>4).
DEVI v16bf load_fragB(const __bf16* p) {
  FragU f;
  f.u[0] = *reinterpret_cast<const uint4*>(p);
  f.u[1] = *reinterpret_cast<const uint4*>(p + 16);
  return f.v;
}
DEVI v8f wmma_bf16(v16bf a, v16bf b, v8f c) {
  return __builtin_amdgcn_wmma_f32_16x16x32_bf16(false, a, false, b, (short)0, c, false, false);
}

// ---------------------------------------------------------------------------
// Generic GEMM:  out[M,N] = A[M,K] * Bt[N,K]^T  (bf16 in, f32 accumulate)
// EPI: 0 -> f32, 1 -> bf16, 2 -> gelu->bf16, 3 -> +R(bf16)->bf16, 4 -> +R(bf16)->f32
// Block tile 128x128, 8 waves, each wave 2 M-tiles x 4 N-tiles.
// ---------------------------------------------------------------------------
template <int EPI>
__global__ __launch_bounds__(256) void gemm_bf16_k(
    const __bf16* __restrict__ A, const __bf16* __restrict__ Bt,
    const __bf16* __restrict__ R, float* __restrict__ outF,
    __bf16* __restrict__ outB, int M, int N, int K)
{
  const int w = threadIdx.x >> 5, lane = threadIdx.x & 31;
  const int lr = lane & 15, lh = lane >> 4;
  const int m0 = blockIdx.x * 128 + (w & 3) * 32;
  const int n0 = blockIdx.y * 128 + (w >> 2) * 64;
  v8f acc[2][4] = {};
  for (int kk = 0; kk < K; kk += 32) {
    v16bf a[2], b[4];
#pragma unroll
    for (int i = 0; i < 2; ++i)
      a[i] = load_fragA(A + (size_t)(m0 + i * 16 + lr) * K + kk + 8 * lh);
#pragma unroll
    for (int j = 0; j < 4; ++j)
      b[j] = load_fragB(Bt + (size_t)(n0 + j * 16 + lr) * K + kk + 8 * lh);
#pragma unroll
    for (int i = 0; i < 2; ++i)
#pragma unroll
      for (int j = 0; j < 4; ++j)
        acc[i][j] = wmma_bf16(a[i], b[j], acc[i][j]);
  }
#pragma unroll
  for (int i = 0; i < 2; ++i) {
#pragma unroll
    for (int j = 0; j < 4; ++j) {
      const int col  = n0 + j * 16 + lr;
      const int row0 = m0 + i * 16 + lh * 8;
#pragma unroll
      for (int g = 0; g < 8; ++g) {
        const size_t idx = (size_t)(row0 + g) * N + col;
        float v = acc[i][j][g];
        if (EPI == 3 || EPI == 4) v += bf2f(R[idx]);
        if (EPI == 2) v = geluf(v);
        if (EPI == 0 || EPI == 4) outF[idx] = v;
        else                      outB[idx] = f2bf(v);
      }
    }
  }
}

// ---------------------------------------------------------------------------
// Conv3x3 SAME, 128->128 ch over an 11x11 grid, one agent per block.
// Implicit im2col GEMM: M=pixels(121->128), N=co(128), K=9*128.
// Input staged in LDS with zero halo [13][13][128] bf16.
// MODE 0: out = gelu(conv(in));  MODE 1: out = gelu(conv(in) + res)
// ---------------------------------------------------------------------------
template <int MODE>
__global__ __launch_bounds__(256) void conv3x3_k(
    const __bf16* __restrict__ in, const __bf16* __restrict__ Wt,
    const __bf16* __restrict__ res, __bf16* __restrict__ out)
{
  __shared__ __bf16 lds[13 * 13 * 128];   // 43,264 B
  const int n = blockIdx.x, tid = threadIdx.x;
  uint4* l4 = reinterpret_cast<uint4*>(lds);
  const uint4 z = make_uint4(0, 0, 0, 0);
  for (int i = tid; i < 13 * 13 * 16; i += 256) l4[i] = z;
  __syncthreads();
  const uint4* g4 = reinterpret_cast<const uint4*>(in + (size_t)n * 121 * 128);
  for (int i = tid; i < 121 * 16; i += 256) {
    const int p = i >> 4, c = i & 15;
    const int py = p / 11, px = p - py * 11;
    l4[((py + 1) * 13 + (px + 1)) * 16 + c] = g4[i];
  }
  __syncthreads();

  const int w = tid >> 5, lane = tid & 31;
  const int lr = lane & 15, lh = lane >> 4;
  const int wm = w & 3, wn = w >> 2;
  int abase[2];
#pragma unroll
  for (int t = 0; t < 2; ++t) {
    int p = wm * 32 + t * 16 + lr;
    if (p > 120) p = 120;                 // pad rows compute garbage, never stored
    const int py = p / 11, px = p - py * 11;
    abase[t] = (py * 13 + px) * 128;
  }
  v8f acc[2][4] = {};
  for (int kp = 0; kp < 9; ++kp) {
    const int sh = ((kp / 3) * 13 + (kp % 3)) * 128;
#pragma unroll
    for (int cb = 0; cb < 4; ++cb) {
      const int kk = kp * 128 + cb * 32;
      v16bf a[2], b[4];
#pragma unroll
      for (int t = 0; t < 2; ++t)
        a[t] = load_fragA(lds + abase[t] + sh + cb * 32 + 8 * lh);
#pragma unroll
      for (int t = 0; t < 4; ++t)
        b[t] = load_fragB(Wt + (size_t)(wn * 64 + t * 16 + lr) * 1152 + kk + 8 * lh);
#pragma unroll
      for (int i = 0; i < 2; ++i)
#pragma unroll
        for (int j = 0; j < 4; ++j)
          acc[i][j] = wmma_bf16(a[i], b[j], acc[i][j]);
    }
  }
#pragma unroll
  for (int i = 0; i < 2; ++i) {
#pragma unroll
    for (int j = 0; j < 4; ++j) {
      const int co = wn * 64 + j * 16 + lr;
      const int p0 = wm * 32 + i * 16 + lh * 8;
#pragma unroll
      for (int g = 0; g < 8; ++g) {
        const int p = p0 + g;
        if (p < 121) {
          const size_t idx = ((size_t)n * 121 + p) * 128 + co;
          float v = acc[i][j][g];
          if (MODE == 1) v += bf2f(res[idx]);
          out[idx] = f2bf(geluf(v));
        }
      }
    }
  }
}

// ------------------------------- small kernels -----------------------------
__global__ void cast_bf16_k(const float* __restrict__ in, __bf16* __restrict__ out, int n) {
  const int i = blockIdx.x * 256 + threadIdx.x;
  if (i < n) out[i] = f2bf(in[i]);
}
// f32 [K][N] row-major  ->  bf16 [N][K]
__global__ void repack_t_k(const float* __restrict__ in, __bf16* __restrict__ out, int K, int N) {
  const int i = blockIdx.x * 256 + threadIdx.x;
  if (i < K * N) {
    const int nn = i / K, k = i - nn * K;
    out[i] = f2bf(in[(size_t)k * N + nn]);
  }
}
// f32 [co][ci][3][3] -> bf16 [co][kpos*128+ci]
__global__ void repack_conv_k(const float* __restrict__ in, __bf16* __restrict__ out) {
  const int i = blockIdx.x * 256 + threadIdx.x;
  if (i < 128 * 1152) {
    const int co = i / 1152, r = i - co * 1152;
    const int kp = r >> 7, ci = r & 127;
    const int ky = kp / 3, kx = kp - ky * 3;
    out[i] = f2bf(in[(((size_t)co * 128 + ci) * 3 + ky) * 3 + kx]);
  }
}
__global__ void embed_k(const int* __restrict__ obs, const __bf16* __restrict__ tok,
                        __bf16* __restrict__ out) {
  const int i = blockIdx.x * 256 + threadIdx.x;
  if (i < 4096 * 121 * 128) {
    const int ci = i & 127, np = i >> 7;
    out[i] = tok[(size_t)obs[np] * 128 + ci];
  }
}
__global__ void pool_k(const __bf16* __restrict__ sp, __bf16* __restrict__ pooled) {
  const int n = blockIdx.x, ci = threadIdx.x;
  float s = 0.f;
  for (int p = 0; p < 121; ++p) s += bf2f(sp[((size_t)n * 121 + p) * 128 + ci]);
  pooled[(size_t)n * 128 + ci] = f2bf(s * (1.f / 121.f));
}
// Me[2][1024] = We[2][256] @ Wedge[256][1024]
__global__ void me_k(const float* __restrict__ We, const float* __restrict__ Wedge,
                     float* __restrict__ Me) {
  const int i = blockIdx.x * 256 + threadIdx.x;
  if (i < 2048) {
    const int j = i >> 10, m = i & 1023;
    float s = 0.f;
    for (int t = 0; t < 256; ++t) s += We[j * 256 + t] * Wedge[(size_t)t * 1024 + m];
    Me[i] = s;
  }
}
__global__ void rmsnorm_k(const float* __restrict__ in, const float* __restrict__ w,
                          float* __restrict__ outF, __bf16* __restrict__ outB, int D) {
  const int row = blockIdx.x * 4 + (threadIdx.x >> 5);
  const int lane = threadIdx.x & 31;
  const float* r = in + (size_t)row * D;
  float ss = 0.f;
  for (int d = lane; d < D; d += 32) { const float v = r[d]; ss += v * v; }
  for (int o = 16; o > 0; o >>= 1) ss += __shfl_xor(ss, o, 32);
  const float inv = rsqrtf(ss / (float)D + 1e-5f);
  for (int d = lane; d < D; d += 32) {
    const float o = r[d] * inv * w[d];
    if (outF) outF[(size_t)row * D + d] = o;
    if (outB) outB[(size_t)row * D + d] = f2bf(o);
  }
}
// ------------------------------- graph CSR build ---------------------------
__global__ void zero_i_k(int* p, int n) {
  const int i = blockIdx.x * 256 + threadIdx.x;
  if (i < n) p[i] = 0;
}
__global__ void edge_count_k(const int* __restrict__ chat, int* __restrict__ cnt,
                             int* __restrict__ edst) {
  const int e = blockIdx.x * 256 + threadIdx.x;
  if (e < 32768) {
    const int i = e >> 3;
    const int dst = (i & ~127) + chat[e];  // (i/C)*C + nb, C=128
    edst[e] = dst;
    atomicAdd(&cnt[dst], 1);
  }
}
__global__ void scan_k(const int* __restrict__ cnt, int* __restrict__ offs) {
  __shared__ int part[256];
  const int t = threadIdx.x;
  int s = 0;
  for (int u = 0; u < 16; ++u) s += cnt[t * 16 + u];
  part[t] = s;
  __syncthreads();
  if (t == 0) {
    int run = 0;
    for (int u = 0; u < 256; ++u) { const int tmp = part[u]; part[u] = run; run += tmp; }
  }
  __syncthreads();
  int run = part[t];
  for (int u = 0; u < 16; ++u) { offs[t * 16 + u] = run; run += cnt[t * 16 + u]; }
}
__global__ void edge_fill_k(const int* __restrict__ edst, const int* __restrict__ offs,
                            int* __restrict__ fill, int* __restrict__ elist) {
  const int e = blockIdx.x * 256 + threadIdx.x;
  if (e < 32768) {
    const int dst = edst[e];
    const int pos = offs[dst] + atomicAdd(&fill[dst], 1);
    elist[pos] = e;
  }
}
// ------------------------------- edge attention ----------------------------
// One block per dst node, one wave per head (H=4, d=256). Online softmax over
// self edge + variable in-degree CSR list. ee folded in via Me = We@Wedge.
__global__ __launch_bounds__(128) void attn_k(
    const __bf16* __restrict__ q, const __bf16* __restrict__ k, const __bf16* __restrict__ v,
    const float* __restrict__ Me, const int* __restrict__ offs, const int* __restrict__ cnt,
    const int* __restrict__ elist, const float* __restrict__ coords,
    __bf16* __restrict__ agg)
{
  const int j = blockIdx.x;
  const int h = threadIdx.x >> 5, lane = threadIdx.x & 31;
  const int d0 = h * 256 + lane * 8;
  float qv[8], me0[8], me1[8];
#pragma unroll
  for (int t = 0; t < 8; ++t) {
    qv[t]  = bf2f(q[(size_t)j * 1024 + d0 + t]);
    me0[t] = Me[d0 + t];
    me1[t] = Me[1024 + d0 + t];
  }
  float s = 0.f;
#pragma unroll
  for (int t = 0; t < 8; ++t) s += qv[t] * bf2f(k[(size_t)j * 1024 + d0 + t]);
  for (int o = 16; o > 0; o >>= 1) s += __shfl_xor(s, o, 32);
  float m = s * 0.0625f;   // 1/sqrt(256)
  float den = 1.f;
  float acc[8];
#pragma unroll
  for (int t = 0; t < 8; ++t) acc[t] = bf2f(v[(size_t)j * 1024 + d0 + t]);

  const int beg = offs[j], num = cnt[j];
  for (int eidx = 0; eidx < num; ++eidx) {
    const int e = elist[beg + eidx];
    const int i = e >> 3, l = e & 7;
    const float c0 = coords[i * 16 + 2 * l];
    const float c1 = coords[i * 16 + 2 * l + 1];
    float sc = 0.f;
#pragma unroll
    for (int t = 0; t < 8; ++t)
      sc += qv[t] * (bf2f(k[(size_t)i * 1024 + d0 + t]) + c0 * me0[t] + c1 * me1[t]);
    for (int o = 16; o > 0; o >>= 1) sc += __shfl_xor(sc, o, 32);
    sc *= 0.0625f;
    const float mn = fmaxf(m, sc);
    const float eo = __expf(m - mn), en = __expf(sc - mn);
#pragma unroll
    for (int t = 0; t < 8; ++t)
      acc[t] = acc[t] * eo + en * (bf2f(v[(size_t)i * 1024 + d0 + t]) + c0 * me0[t] + c1 * me1[t]);
    den = den * eo + en;
    m = mn;
  }
  const float r = 1.f / den;
#pragma unroll
  for (int t = 0; t < 8; ++t) agg[(size_t)j * 1024 + d0 + t] = f2bf(acc[t] * r);
}

// ---------------------------------------------------------------------------
#define G1(n) dim3(((n) + 255) / 256), dim3(256), 0, stream

extern "C" void kernel_launch(void* const* d_in, const int* in_sizes, int n_in,
                              void* d_out, int out_size, void* d_ws, size_t ws_size,
                              hipStream_t stream)
{
  (void)in_sizes; (void)n_in; (void)out_size; (void)ws_size;
  // jax tree_flatten order (dict keys sorted):
  // 0 observations, 1 agent_chat_ids, 2 agents_rel_coords,
  // 3 W_action, 4 W_latent,
  // 5..12 dec_gnn[0]{We,Wedge,Wk,Wout,Wq,Wskip,Wv,ln}, 13..20 dec_gnn[1],
  // 21..28 enc_gnn[0], 29..36 enc_gnn[1],
  // 37..39 enc_mlp[0]{W1,W2,ln}, 40..42 enc_mlp[1],
  // 43 ln_action, 44 ln_spatial, 45..48 resnet w1/w2 x2, 49 tok_emb
  const int*   obs      = (const int*)d_in[0];
  const int*   chat     = (const int*)d_in[1];
  const float* coords   = (const float*)d_in[2];
  const float* W_action = (const float*)d_in[3];
  const float* W_latent = (const float*)d_in[4];
  const float* ln_act   = (const float*)d_in[43];
  const float* ln_sp    = (const float*)d_in[44];
  const float* convw[4] = {(const float*)d_in[45], (const float*)d_in[46],
                           (const float*)d_in[47], (const float*)d_in[48]};
  const float* tok_emb  = (const float*)d_in[49];
  const int gbase[4] = {21, 29, 5, 13};   // processing order: enc0, enc1, dec0, dec1

  char* ws = (char*)d_ws;
  const size_t SPB = (size_t)4096 * 121 * 128 * 2;   // 126,877,696 B per spatial buffer
  __bf16* sp0 = (__bf16*)ws;
  __bf16* sp1 = (__bf16*)(ws + SPB);
  char* wr = ws + 2 * SPB;                           // weight/meta region (~13.2 MB)
  __bf16* convWt[4];
  for (int i = 0; i < 4; ++i) convWt[i] = (__bf16*)(wr + (size_t)i * 294912);
  __bf16* tokbf = (__bf16*)(wr + 1179648);
  __bf16* WlatT = (__bf16*)(wr + 1212416);
  const size_t g0 = 1277952, gst = 2621440;
  __bf16 *WqT[4], *WkT[4], *WvT[4], *WskT[4], *WoT[4];
  for (int l = 0; l < 4; ++l) {
    char* b = wr + g0 + (size_t)l * gst;
    WqT[l]  = (__bf16*)(b + 0);
    WkT[l]  = (__bf16*)(b + 524288);
    WvT[l]  = (__bf16*)(b + 1048576);
    WskT[l] = (__bf16*)(b + 1572864);
    WoT[l]  = (__bf16*)(b + 2097152);
  }
  const size_t mm0 = g0 + 4 * gst;                   // 11,763,712
  __bf16 *W1T[2], *W2T[2];
  for (int i = 0; i < 2; ++i) {
    W1T[i] = (__bf16*)(wr + mm0 + (size_t)i * 524288);
    W2T[i] = (__bf16*)(wr + mm0 + (size_t)i * 524288 + 262144);
  }
  const size_t wact = mm0 + 1048576;                 // 12,812,288
  __bf16* WactT = (__bf16*)(wr + wact);
  float*  Me    = (float*)(wr + wact + 65536);
  char* gb = wr + wact + 65536 + 8192;
  int* cnt   = (int*)(gb);
  int* offs  = (int*)(gb + 16384);
  int* fill  = (int*)(gb + 32768);
  int* edst  = (int*)(gb + 49152);
  int* elist = (int*)(gb + 180224);
  // activations: graph-phase buffers alias sp0 region; x-phase aliases sp1 region
  __bf16* qb  = (__bf16*)(ws + 0);
  __bf16* kb  = (__bf16*)(ws + 8388608);
  __bf16* vb  = (__bf16*)(ws + 16777216);
  __bf16* agg = (__bf16*)(ws + 25165824);
  __bf16* cat = (__bf16*)(ws + 33554432);
  __bf16* hb  = (__bf16*)(ws + 41943040);
  __bf16* pooled = (__bf16*)(ws + SPB);
  float*  xf32   = (float*)(ws + SPB + 2097152);
  __bf16* x_bf   = (__bf16*)(ws + SPB + 8388608);

  // ---- weight prep (bf16, B pre-transposed to [N][K]) ----
  cast_bf16_k<<<G1(16384)>>>(tok_emb, tokbf, 16384);
  for (int i = 0; i < 4; ++i) repack_conv_k<<<G1(128 * 1152)>>>(convw[i], convWt[i]);
  repack_t_k<<<G1(128 * 256)>>>(W_latent, WlatT, 128, 256);
  for (int l = 0; l < 4; ++l) {
    const int b = gbase[l];
    repack_t_k<<<G1(256 * 1024)>>>((const float*)d_in[b + 4], WqT[l], 256, 1024);
    repack_t_k<<<G1(256 * 1024)>>>((const float*)d_in[b + 2], WkT[l], 256, 1024);
    repack_t_k<<<G1(256 * 1024)>>>((const float*)d_in[b + 6], WvT[l], 256, 1024);
    repack_t_k<<<G1(256 * 1024)>>>((const float*)d_in[b + 5], WskT[l], 256, 1024);
    repack_t_k<<<G1(1024 * 256)>>>((const float*)d_in[b + 3], WoT[l], 1024, 256);
  }
  for (int i = 0; i < 2; ++i) {
    const int b = (i == 0) ? 37 : 40;
    repack_t_k<<<G1(256 * 512)>>>((const float*)d_in[b + 0], W1T[i], 256, 512);
    repack_t_k<<<G1(512 * 256)>>>((const float*)d_in[b + 1], W2T[i], 512, 256);
  }
  repack_t_k<<<G1(256 * 128)>>>(W_action, WactT, 256, 128);

  // ---- graph CSR build (static across layers) ----
  zero_i_k<<<G1(12288)>>>(cnt, 12288);               // cnt + offs + fill contiguous
  edge_count_k<<<G1(32768)>>>(chat, cnt, edst);
  scan_k<<<1, 256, 0, stream>>>(cnt, offs);
  edge_fill_k<<<G1(32768)>>>(edst, offs, fill, elist);

  // ---- spatial encoder ----
  embed_k<<<G1(4096 * 121 * 128)>>>(obs, tokbf, sp0);
  conv3x3_k<0><<<4096, 256, 0, stream>>>(sp0, convWt[0], nullptr, sp1);
  conv3x3_k<1><<<4096, 256, 0, stream>>>(sp1, convWt[1], sp0, sp0);
  conv3x3_k<0><<<4096, 256, 0, stream>>>(sp0, convWt[2], nullptr, sp1);
  conv3x3_k<1><<<4096, 256, 0, stream>>>(sp1, convWt[3], sp0, sp0);
  pool_k<<<4096, 128, 0, stream>>>(sp0, pooled);
  gemm_bf16_k<0><<<dim3(32, 2), 256, 0, stream>>>(pooled, WlatT, nullptr, xf32, nullptr, 4096, 256, 128);
  rmsnorm_k<<<1024, 128, 0, stream>>>(xf32, ln_sp, nullptr, x_bf, 256);

  // ---- graph layers ----
  auto run_gnn = [&](int b, int l) {
    me_k<<<8, 256, 0, stream>>>((const float*)d_in[b + 0], (const float*)d_in[b + 1], Me);
    gemm_bf16_k<1><<<dim3(32, 8), 256, 0, stream>>>(x_bf, WqT[l], nullptr, nullptr, qb, 4096, 1024, 256);
    gemm_bf16_k<1><<<dim3(32, 8), 256, 0, stream>>>(x_bf, WkT[l], nullptr, nullptr, kb, 4096, 1024, 256);
    gemm_bf16_k<1><<<dim3(32, 8), 256, 0, stream>>>(x_bf, WvT[l], nullptr, nullptr, vb, 4096, 1024, 256);
    attn_k<<<4096, 128, 0, stream>>>(qb, kb, vb, Me, offs, cnt, elist, coords, agg);
    gemm_bf16_k<3><<<dim3(32, 8), 256, 0, stream>>>(x_bf, WskT[l], agg, nullptr, cat, 4096, 1024, 256);
    gemm_bf16_k<4><<<dim3(32, 2), 256, 0, stream>>>(cat, WoT[l], x_bf, xf32, nullptr, 4096, 256, 1024);
    rmsnorm_k<<<1024, 128, 0, stream>>>(xf32, (const float*)d_in[b + 7], nullptr, x_bf, 256);
  };
  auto run_mlp = [&](int b, int i) {
    gemm_bf16_k<2><<<dim3(32, 4), 256, 0, stream>>>(x_bf, W1T[i], nullptr, nullptr, hb, 4096, 512, 256);
    gemm_bf16_k<4><<<dim3(32, 2), 256, 0, stream>>>(hb, W2T[i], x_bf, xf32, nullptr, 4096, 256, 512);
    rmsnorm_k<<<1024, 128, 0, stream>>>(xf32, (const float*)d_in[b + 2], nullptr, x_bf, 256);
  };
  run_gnn(21, 0); run_mlp(37, 0);
  run_gnn(29, 1); run_mlp(40, 1);
  run_gnn(5, 2);
  run_gnn(13, 3);

  // ---- action head ----
  gemm_bf16_k<0><<<dim3(32, 1), 256, 0, stream>>>(x_bf, WactT, nullptr, xf32, nullptr, 4096, 128, 256);
  rmsnorm_k<<<1024, 128, 0, stream>>>(xf32, ln_act, (float*)d_out, nullptr, 128);
}
